// AnchorUpdate_56023553409077
// MI455X (gfx1250) — matrix operand order repaired and checked by reading, not compiled, not run on hardware
//
#include <hip/hip_runtime.h>
#include <hip/hip_bf16.h>
#include <math.h>

// ---------------------------------------------------------------------------
// Problem constants (from reference): B=8, N=384, K=96, H=128, E=16
// ---------------------------------------------------------------------------
#define B_  8
#define N_  384
#define K_  96
#define H_  128
#define BN_ (B_ * N_)   // 3072
#define BK_ (B_ * K_)   // 768
#define CN_ 64          // (b,n) groups per a2n chunk -> 6144 pair rows / chunk
#define EPS_ 1e-8f
#define INF_ 1000000.0f

typedef __attribute__((ext_vector_type(16))) __bf16 v16bf;
typedef __attribute__((ext_vector_type(8)))  float  f32x8;

// ---------------------------------------------------------------------------
// WMMA GEMM:  C[r,c] = act( sum_k A[r,k] * Wt[c,k] + bias[c] )
//   A   : f32 row-major (rows x Kd), lda = Kd   (Kd multiple of 32)
//   Wt  : bf16 row-major (Co x Kd)  (i.e. W transposed, pre-converted)
//   C   : f32 row-major (rows x Co)
// One wave computes a 32 x (16*NT) tile: two A fragments share each B
// fragment -> 2*NT v_wmma_f32_16x16x32_bf16 per 32-wide k step.  All NT B
// tiles are loaded up-front each k step (one load clause) so the WMMAs can
// overlap outstanding loads instead of waiting per tile.  rows must be a
// multiple of 32.  Batched via grid.z with element strides sA/sW/sC.
// ---------------------------------------------------------------------------
template<int NT, bool RELU>
__global__ void gemm_bf16_wmma(const float* __restrict__ A,
                               const __bf16* __restrict__ Wt,
                               const float* __restrict__ bias,
                               float* __restrict__ C,
                               int rows, int Co, int Kd, int colBase,
                               long sA, long sW, long sC)
{
    const int bz = blockIdx.z;
    A  += (size_t)bz * sA;
    Wt += (size_t)bz * sW;
    C  += (size_t)bz * sC;

    const int wave = threadIdx.x >> 5;
    const int lane = threadIdx.x & 31;
    const int row0 = (blockIdx.x * 4 + wave) * 32;
    if (row0 >= rows) return;                       // wave-uniform
    const int col0 = colBase + blockIdx.y * (16 * NT);
    const int m    = lane & 15;
    const int khi  = lane >> 4;                     // 0: lanes 0-15, 1: 16-31

    f32x8 acc0[NT], acc1[NT];
#pragma unroll
    for (int t = 0; t < NT; ++t) {
        float bv = bias ? bias[col0 + 16 * t + m] : 0.0f;
#pragma unroll
        for (int j = 0; j < 8; ++j) { acc0[t][j] = bv; acc1[t][j] = bv; }
    }

    // A lane base: row = row0+m (+16) ; K offsets khi*16 + [0..15]
    const float* arow0 = A + (size_t)(row0 + m) * Kd + khi * 16;
    const float* arow1 = arow0 + (size_t)16 * Kd;
    // B lane base: n = m ; K offsets khi*16 + [0..15]
    const __bf16* wrow = Wt + (size_t)(khi * 16) + (size_t)(col0 + m) * Kd;

    for (int kk = 0; kk < Kd; kk += 32) {
        // A fragments (two row strips)
        f32x8 a0 = *(const f32x8*)(arow0 + kk);
        f32x8 a1 = *(const f32x8*)(arow0 + kk + 8);
        f32x8 a2 = *(const f32x8*)(arow1 + kk);
        f32x8 a3 = *(const f32x8*)(arow1 + kk + 8);
        // All NT B fragments up-front (single load clause)
        v16bf bfr[NT];
#pragma unroll
        for (int t = 0; t < NT; ++t)
            bfr[t] = *(const v16bf*)(wrow + (size_t)t * 16 * Kd + kk);   // 32B aligned
        v16bf af0, af1;
#pragma unroll
        for (int j = 0; j < 8; ++j) {
            af0[j]     = (__bf16)a0[j];
            af0[8 + j] = (__bf16)a1[j];
            af1[j]     = (__bf16)a2[j];
            af1[8 + j] = (__bf16)a3[j];
        }
#pragma unroll
        for (int t = 0; t < NT; ++t) {
            acc0[t] = __builtin_amdgcn_wmma_f32_16x16x32_bf16(
                false, af0, false, bfr[t], (short)0, acc0[t], false, false);
            acc1[t] = __builtin_amdgcn_wmma_f32_16x16x32_bf16(
                false, af1, false, bfr[t], (short)0, acc1[t], false, false);
        }
    }

    // C layout: VGPR j -> row = row0 + khi*8 + j (strip0), +16 (strip1)
#pragma unroll
    for (int t = 0; t < NT; ++t) {
        int c = col0 + 16 * t + m;
        float* crow0 = C + (size_t)(row0 + khi * 8) * Co + c;
        float* crow1 = crow0 + (size_t)16 * Co;
#pragma unroll
        for (int j = 0; j < 8; ++j) {
            float v0 = acc0[t][j], v1 = acc1[t][j];
            if (RELU) { v0 = v0 > 0.0f ? v0 : 0.0f; v1 = v1 > 0.0f ? v1 : 0.0f; }
            crow0[(size_t)j * Co] = v0;
            crow1[(size_t)j * Co] = v1;
        }
    }
}

// ---------------------------------------------------------------------------
// Small scalar GEMM for E=16-dim edge projections.
//   wTrans==0: C[r,c] (+)= sum_k A[r,k] * W[k*ldw + c]
//   wTrans==1: C[r,c] (+)= sum_k A[r,k] * W[c*ldw + k]
// ---------------------------------------------------------------------------
__global__ void sgemm_naive(const float* __restrict__ A, const float* __restrict__ W,
                            float* __restrict__ C, int rows, int Co, int Kd,
                            int ldw, int wTrans, int accum)
{
    int i = blockIdx.x * blockDim.x + threadIdx.x;
    if (i >= rows * Co) return;
    int r = i / Co, c = i % Co;
    float acc = 0.0f;
    for (int k = 0; k < Kd; ++k) {
        float w = wTrans ? W[(size_t)c * ldw + k] : W[(size_t)k * ldw + c];
        acc += A[(size_t)r * Kd + k] * w;
    }
    if (accum) C[i] += acc; else C[i] = acc;
}

// Wt[c*Kd + k] = (bf16) W[k*ldw + c]
__global__ void conv_transpose_bf16(const float* __restrict__ W, __bf16* __restrict__ Wt,
                                    int Kd, int Co, int ldw)
{
    int i = blockIdx.x * blockDim.x + threadIdx.x;
    if (i >= Kd * Co) return;
    int c = i / Kd, k = i % Kd;
    Wt[i] = (__bf16)W[(size_t)k * ldw + c];
}

// Y[r*cols + c] = (bf16) X[r*ldx + off + c]
__global__ void conv_row_bf16(const float* __restrict__ X, __bf16* __restrict__ Y,
                              int rows, int cols, int ldx, int off)
{
    int i = blockIdx.x * blockDim.x + threadIdx.x;
    if (i >= rows * cols) return;
    int r = i / cols, c = i % cols;
    Y[i] = (__bf16)X[(size_t)r * ldx + off + c];
}

// Batched activation transpose:  Y[b][c*nIdx + n] = (bf16) X[b][n*ldx + off + c]
__global__ void conv_tr_bf16(const float* __restrict__ X, __bf16* __restrict__ Y,
                             int nIdx, int cIdx, int ldx, int off)
{
    int b = blockIdx.y;
    int i = blockIdx.x * blockDim.x + threadIdx.x;
    if (i >= nIdx * cIdx) return;
    int c = i / nIdx, n = i % nIdx;
    Y[(size_t)b * cIdx * nIdx + i] =
        (__bf16)X[(size_t)b * nIdx * ldx + (size_t)n * ldx + off + c];
}

// y[r,h] = x[r,h] * mask[r]
__global__ void maskmul_kernel(const float* __restrict__ x, const float* __restrict__ m,
                               float* __restrict__ y, int rows, int cols)
{
    int i = blockIdx.x * blockDim.x + threadIdx.x;
    if (i >= rows * cols) return;
    y[i] = x[i] * m[i / cols];
}

// score[r] = tanh( dot(sv[r,:], w) / ||w|| )
__global__ void score_kernel(const float* __restrict__ sv, const float* __restrict__ w,
                             float* __restrict__ score, int rows)
{
    int r = blockIdx.x * blockDim.x + threadIdx.x;
    if (r >= rows) return;
    float nrm = 0.0f, dot = 0.0f;
    for (int h = 0; h < H_; ++h) { nrm += w[h] * w[h]; dot += sv[(size_t)r * H_ + h] * w[h]; }
    score[r] = tanhf(dot / sqrtf(nrm));
}

// Iterative top-K per batch (stable: ties -> lowest index, like lax.top_k)
__global__ void topk_kernel(const float* __restrict__ score, int* __restrict__ topi,
                            float* __restrict__ tops)
{
    __shared__ float sv[N_];
    __shared__ float rv[128];
    __shared__ int   ri[128];
    int b = blockIdx.x, tid = threadIdx.x;
    for (int n = tid; n < N_; n += 128) sv[n] = score[b * N_ + n];
    __syncthreads();
    for (int it = 0; it < K_; ++it) {
        float bv = -3.0e38f; int bi = 0;
        for (int n = tid; n < N_; n += 128) {
            float v = sv[n];
            if (v > bv) { bv = v; bi = n; }
        }
        rv[tid] = bv; ri[tid] = bi;
        __syncthreads();
        for (int s = 64; s > 0; s >>= 1) {
            if (tid < s) {
                if (rv[tid + s] > rv[tid] ||
                    (rv[tid + s] == rv[tid] && ri[tid + s] < ri[tid])) {
                    rv[tid] = rv[tid + s]; ri[tid] = ri[tid + s];
                }
            }
            __syncthreads();
        }
        if (tid == 0) {
            topi[b * K_ + it] = ri[0];
            tops[b * K_ + it] = rv[0];
            sv[ri[0]] = -3.0e38f;
        }
        __syncthreads();
    }
}

// anchor_x[bk,:] = node_x[b*N+idx,:];  anchor_f[bk,h] = sv[b*N+idx,h] * tops[bk]
__global__ void gather_anchor(const float* __restrict__ nx, const float* __restrict__ sv,
                              const int* __restrict__ topi, const float* __restrict__ tops,
                              float* __restrict__ ax, float* __restrict__ af)
{
    int i = blockIdx.x * blockDim.x + threadIdx.x;
    if (i >= BK_ * H_) return;
    int bk = i / H_, h = i % H_;
    int b = bk / K_;
    int idx = topi[bk];
    size_t src = (size_t)b * N_ + idx;
    af[i] = sv[src * H_ + h] * tops[bk];
    if (h < 3) ax[(size_t)bk * 3 + h] = nx[src * 3 + h];
}

// Per (b,q): add edge-RBF term to logits, apply multiplicative mask, softmax,
// write attn probs back to L and edge-weighted attn sums se[bq,16].
__global__ void attn_softmax_kernel(const float* __restrict__ ax, const float* __restrict__ kx,
                                    const float* __restrict__ tvec, float* __restrict__ L,
                                    const float* __restrict__ mask, float* __restrict__ se,
                                    int Kv)
{
    __shared__ float sl[N_];
    __shared__ float sd[N_];
    __shared__ float red[128];
    __shared__ float sacc[16];
    int bq = blockIdx.x;
    int b  = bq / K_;
    int tid = threadIdx.x;
    float qx = ax[bq * 3 + 0], qy = ax[bq * 3 + 1], qz = ax[bq * 3 + 2];
    float tq[16];
#pragma unroll
    for (int e = 0; e < 16; ++e) tq[e] = tvec[(size_t)bq * 16 + e];

    for (int n = tid; n < Kv; n += 128) {
        const float* kp = kx + ((size_t)b * Kv + n) * 3;
        float dx = qx - kp[0] + EPS_;
        float dy = qy - kp[1] + EPS_;
        float dz = qz - kp[2] + EPS_;
        float d = sqrtf(dx * dx + dy * dy + dz * dz) * 0.1f;
        sd[n] = d;
        float logit = L[(size_t)bq * Kv + n];
#pragma unroll
        for (int e = 0; e < 16; ++e) {
            float u = (d - 1.33333333f * e) * 0.8f;   // (d - mu_e)/sigma, sigma=1.25
            logit += expf(-u * u) * tq[e];
        }
        float mk = mask ? mask[(size_t)b * Kv + n] : 1.0f;
        logit *= (mk - 1.0f) * INF_;                  // faithful to reference
        sl[n] = logit;
    }
    __syncthreads();
    float mx = -3.0e38f;
    for (int n = tid; n < Kv; n += 128) mx = fmaxf(mx, sl[n]);
    red[tid] = mx; __syncthreads();
    for (int s = 64; s > 0; s >>= 1) { if (tid < s) red[tid] = fmaxf(red[tid], red[tid + s]); __syncthreads(); }
    mx = red[0]; __syncthreads();
    float sm = 0.0f;
    for (int n = tid; n < Kv; n += 128) { float e_ = expf(sl[n] - mx); sl[n] = e_; sm += e_; }
    red[tid] = sm; __syncthreads();
    for (int s = 64; s > 0; s >>= 1) { if (tid < s) red[tid] += red[tid + s]; __syncthreads(); }
    float inv = 1.0f / red[0];
    if (tid < 16) sacc[tid] = 0.0f;
    __syncthreads();
    for (int n = tid; n < Kv; n += 128) {
        float a = sl[n] * inv;
        L[(size_t)bq * Kv + n] = a;
        float d = sd[n];
#pragma unroll
        for (int e = 0; e < 16; ++e) {
            float u = (d - 1.33333333f * e) * 0.8f;
            atomicAdd(&sacc[e], a * expf(-u * u));
        }
    }
    __syncthreads();
    if (tid < 16) se[(size_t)bq * 16 + tid] = sacc[tid];
}

// out[r,:] = LN( a[r,:] + b[r,:]*mask[r] ) * g + beta   (b, mask optional)
__global__ void ln_kernel(const float* __restrict__ a, const float* __restrict__ b,
                          const float* __restrict__ mask, const float* __restrict__ g,
                          const float* __restrict__ bet, float* __restrict__ out)
{
    int r = blockIdx.x, h = threadIdx.x;
    size_t idx = (size_t)r * H_ + h;
    float x = a[idx];
    if (b) { float mv = mask ? mask[r] : 1.0f; x += b[idx] * mv; }
    __shared__ float red[128];
    red[h] = x; __syncthreads();
    for (int s = 64; s > 0; s >>= 1) { if (h < s) red[h] += red[h + s]; __syncthreads(); }
    float mean = red[0] * (1.0f / 128.0f);
    __syncthreads();
    float xm = x - mean;
    red[h] = xm * xm; __syncthreads();
    for (int s = 64; s > 0; s >>= 1) { if (h < s) red[h] += red[h + s]; __syncthreads(); }
    float var = red[0] * (1.0f / 128.0f);
    out[idx] = xm * rsqrtf(var + 1e-5f) * g[h] + bet[h];
}

// a2n pair assembly: h1[local,c] = relu(nbase[bn,c] + abase[b*K+k,c]
//                                       + sum_e edge[e]*We1[e,c] + b1[c])
__global__ void pair_h1_kernel(const float* __restrict__ nbase, const float* __restrict__ abase,
                               const float* __restrict__ nx, const float* __restrict__ ax,
                               const float* __restrict__ We1, const float* __restrict__ b1,
                               float* __restrict__ h1, int bnBase)
{
    __shared__ float eg[16];
    int local = blockIdx.x;
    int bn = bnBase + local / K_;
    int k  = local % K_;
    int b  = bn / N_;
    int tid = threadIdx.x;   // 256
    if (tid < 16) {
        const float* np = nx + (size_t)bn * 3;
        const float* ap = ax + ((size_t)b * K_ + k) * 3;
        float dx = np[0] - ap[0] + EPS_;
        float dy = np[1] - ap[1] + EPS_;
        float dz = np[2] - ap[2] + EPS_;
        float d = sqrtf(dx * dx + dy * dy + dz * dz) * 0.1f;
        float u = (d - 1.33333333f * tid) * 0.8f;
        eg[tid] = expf(-u * u);
    }
    __syncthreads();
    float acc = nbase[(size_t)bn * 256 + tid] + abase[((size_t)b * K_ + k) * 256 + tid] + b1[tid];
#pragma unroll
    for (int e = 0; e < 16; ++e) acc += eg[e] * We1[(size_t)e * 256 + tid];
    h1[(size_t)local * 256 + tid] = acc > 0.0f ? acc : 0.0f;
}

// updN[bnBase+bl, h] = mean_k h3[(bl*K+k), h]
__global__ void mean_over_k(const float* __restrict__ h3, float* __restrict__ updN, int bnBase)
{
    int bl = blockIdx.x, h = threadIdx.x;
    float s = 0.0f;
    for (int k = 0; k < K_; ++k) s += h3[((size_t)bl * K_ + k) * H_ + h];
    updN[(size_t)(bnBase + bl) * H_ + h] = s * (1.0f / K_);
}

__global__ void zero_kernel(float* __restrict__ p, int n)
{
    int i = blockIdx.x * blockDim.x + threadIdx.x;
    if (i < n) p[i] = 0.0f;
}

// ---------------------------------------------------------------------------
// Host side
// ---------------------------------------------------------------------------
// Assumed d_in flatten order: top-level setup_inputs() insertion order
// (node_x, node_features, edge_index, batch, node_mask, params) with the
// params subtree in jax tree_leaves (sorted-key) order:
//   a2a[0](14), a2a[1](14), a2n(16), n2a(14), sl_b1, sl_b2, sl_w1, sl_w2, topk_w
// attn leaf order: kv_b,kv_w,ln1_b,ln1_g,ln2_b,ln2_g, mlp{b1,b2,b3,w1,w2,w3}, q_b,q_w
struct AttnP {
    const float *kv_b, *kv_w, *ln1_b, *ln1_g, *ln2_b, *ln2_g;
    const float *m_b1, *m_b2, *m_b3, *m_w1, *m_w2, *m_w3;
    const float *q_b, *q_w;
};
static AttnP loadAttn(void* const* d, int i) {
    AttnP p;
    p.kv_b = (const float*)d[i + 0];  p.kv_w = (const float*)d[i + 1];
    p.ln1_b = (const float*)d[i + 2]; p.ln1_g = (const float*)d[i + 3];
    p.ln2_b = (const float*)d[i + 4]; p.ln2_g = (const float*)d[i + 5];
    p.m_b1 = (const float*)d[i + 6];  p.m_b2 = (const float*)d[i + 7];
    p.m_b3 = (const float*)d[i + 8];  p.m_w1 = (const float*)d[i + 9];
    p.m_w2 = (const float*)d[i + 10]; p.m_w3 = (const float*)d[i + 11];
    p.q_b = (const float*)d[i + 12];  p.q_w = (const float*)d[i + 13];
    return p;
}
struct AttnW { __bf16 *qwT, *kvfT, *mw1T, *mw2T, *mw3T; };

extern "C" void kernel_launch(void* const* d_in, const int* in_sizes, int n_in,
                              void* d_out, int out_size, void* d_ws, size_t ws_size,
                              hipStream_t stream)
{
    (void)in_sizes; (void)out_size; (void)ws_size;
    if (n_in < 68) return;

    const float* node_x    = (const float*)d_in[0];
    const float* node_f    = (const float*)d_in[1];
    const float* node_mask = (const float*)d_in[4];

    AttnP a2aP0 = loadAttn(d_in, 5);
    AttnP a2aP1 = loadAttn(d_in, 19);
    const float* a2n_ln1_b = (const float*)d_in[33];
    const float* a2n_ln1_g = (const float*)d_in[34];
    const float* a2n_ln2_b = (const float*)d_in[35];
    const float* a2n_ln2_g = (const float*)d_in[36];
    const float* m1_b1 = (const float*)d_in[37];
    const float* m1_b2 = (const float*)d_in[38];
    const float* m1_b3 = (const float*)d_in[39];
    const float* m1_w1 = (const float*)d_in[40];   // 272 x 256
    const float* m1_w2 = (const float*)d_in[41];
    const float* m1_w3 = (const float*)d_in[42];
    const float* m2_b1 = (const float*)d_in[43];
    const float* m2_b2 = (const float*)d_in[44];
    const float* m2_b3 = (const float*)d_in[45];
    const float* m2_w1 = (const float*)d_in[46];
    const float* m2_w2 = (const float*)d_in[47];
    const float* m2_w3 = (const float*)d_in[48];
    AttnP n2aP  = loadAttn(d_in, 49);
    const float* sl_b1  = (const float*)d_in[63];
    const float* sl_b2  = (const float*)d_in[64];
    const float* sl_w1  = (const float*)d_in[65];
    const float* sl_w2  = (const float*)d_in[66];
    const float* topk_w = (const float*)d_in[67];

    // ---- workspace bump allocator (256B aligned) ----
    char* wsp = (char*)d_ws;
    size_t off = 0;
    auto alloc = [&](size_t bytes) -> void* {
        void* p = wsp + off;
        off = (off + bytes + 255) & ~(size_t)255;
        return p;
    };
    auto allocF = [&](size_t n) { return (float*)alloc(n * 4); };
    auto allocH = [&](size_t n) { return (__bf16*)alloc(n * 2); };

    // bf16 weights
    __bf16* slw1T = allocH(H_ * H_);
    __bf16* slw2T = allocH(H_ * H_);
    AttnW aw[3];
    for (int l = 0; l < 3; ++l) {
        aw[l].qwT  = allocH(H_ * H_);
        aw[l].kvfT = allocH(256 * H_);
        aw[l].mw1T = allocH(256 * H_);
        aw[l].mw2T = allocH(256 * 256);
        aw[l].mw3T = allocH(H_ * 256);
    }
    __bf16* WnT   = allocH(256 * H_);
    __bf16* WaT   = allocH(256 * H_);
    __bf16* m1w2T = allocH(256 * 256);
    __bf16* m1w3T = allocH(H_ * 256);
    __bf16* m2w1T = allocH(256 * H_);
    __bf16* m2w2T = allocH(256 * 256);
    __bf16* m2w3T = allocH(H_ * 256);
    __bf16* kbaseB = allocH((size_t)B_ * N_ * H_);
    __bf16* vbaseT = allocH((size_t)B_ * H_ * N_);

    // f32 activations
    float* nfm   = allocF((size_t)BN_ * H_);
    float* hbuf  = allocF((size_t)BN_ * H_);
    float* svec  = allocF((size_t)BN_ * H_);
    float* score = allocF(BN_);
    float* tops  = allocF(BK_);
    int*   topi  = (int*)alloc(BK_ * 4);
    float* anchor_x = allocF((size_t)BK_ * 3);
    float* afA  = allocF((size_t)BK_ * H_);
    float* afBuf = allocF((size_t)BK_ * H_);
    float* qbuf = allocF((size_t)BK_ * H_);
    float* kvbase = allocF((size_t)B_ * N_ * 256);
    float* tvec = allocF((size_t)BK_ * 16);
    float* Lbuf = allocF((size_t)B_ * K_ * N_);
    float* sedge = allocF((size_t)BK_ * 16);
    float* upd  = allocF((size_t)BK_ * H_);
    float* fmid = allocF((size_t)BK_ * H_);
    float* m1b  = allocF((size_t)BK_ * 256);
    float* m2b  = allocF((size_t)BK_ * 256);
    float* m3b  = allocF((size_t)BK_ * H_);
    float* nbase = allocF((size_t)BN_ * 256);
    float* abase = allocF((size_t)BK_ * 256);
    float* h1 = allocF((size_t)CN_ * K_ * 256);
    float* h2 = allocF((size_t)CN_ * K_ * 256);
    float* h3 = allocF((size_t)CN_ * K_ * H_);
    float* updN = allocF((size_t)BN_ * H_);
    float* nf1  = allocF((size_t)BN_ * H_);
    float* nm1  = allocF((size_t)BN_ * 256);
    float* nm2  = allocF((size_t)BN_ * 256);
    float* nm3  = allocF((size_t)BN_ * H_);

    // ---- GEMM dispatch: full 64-wide column blocks (NT=4) + 16*r remainder ----
    auto launch_nt = [&](int NT, bool relu, dim3 grid, const float* A, const __bf16* Wt,
                         const float* bias, float* C, int rows, int Co, int Kd, int colBase,
                         long sA, long sW, long sC) {
        switch (NT) {
        case 4: if (relu) gemm_bf16_wmma<4, true ><<<grid, 128, 0, stream>>>(A, Wt, bias, C, rows, Co, Kd, colBase, sA, sW, sC);
                else      gemm_bf16_wmma<4, false><<<grid, 128, 0, stream>>>(A, Wt, bias, C, rows, Co, Kd, colBase, sA, sW, sC);
                break;
        case 3: if (relu) gemm_bf16_wmma<3, true ><<<grid, 128, 0, stream>>>(A, Wt, bias, C, rows, Co, Kd, colBase, sA, sW, sC);
                else      gemm_bf16_wmma<3, false><<<grid, 128, 0, stream>>>(A, Wt, bias, C, rows, Co, Kd, colBase, sA, sW, sC);
                break;
        case 2: if (relu) gemm_bf16_wmma<2, true ><<<grid, 128, 0, stream>>>(A, Wt, bias, C, rows, Co, Kd, colBase, sA, sW, sC);
                else      gemm_bf16_wmma<2, false><<<grid, 128, 0, stream>>>(A, Wt, bias, C, rows, Co, Kd, colBase, sA, sW, sC);
                break;
        default:if (relu) gemm_bf16_wmma<1, true ><<<grid, 128, 0, stream>>>(A, Wt, bias, C, rows, Co, Kd, colBase, sA, sW, sC);
                else      gemm_bf16_wmma<1, false><<<grid, 128, 0, stream>>>(A, Wt, bias, C, rows, Co, Kd, colBase, sA, sW, sC);
                break;
        }
    };
    auto gemm = [&](const float* A, const __bf16* Wt, const float* bias, float* C,
                    int rows, int Co, int Kd, bool relu,
                    int batches = 1, long sA = 0, long sW = 0, long sC = 0) {
        int gx = (rows + 127) / 128;               // 4 waves x 32 rows per block
        int full = Co / 64, rem = (Co % 64) / 16;  // all Co are multiples of 16
        if (full) launch_nt(4, relu, dim3(gx, full, batches), A, Wt, bias, C, rows, Co, Kd, 0, sA, sW, sC);
        if (rem)  launch_nt(rem, relu, dim3(gx, 1, batches), A, Wt, bias, C, rows, Co, Kd, full * 64, sA, sW, sC);
    };
    auto tconv = [&](const float* W, __bf16* Wt, int Kd, int Co, int ldw) {
        int n = Kd * Co;
        conv_transpose_bf16<<<(n + 255) / 256, 256, 0, stream>>>(W, Wt, Kd, Co, ldw);
    };

    // ---- weight prep ----
    tconv(sl_w1, slw1T, H_, H_, H_);
    tconv(sl_w2, slw2T, H_, H_, H_);
    const AttnP* ps[3] = { &n2aP, &a2aP0, &a2aP1 };
    for (int l = 0; l < 3; ++l) {
        tconv(ps[l]->q_w,  aw[l].qwT,  H_, H_, H_);
        tconv(ps[l]->kv_w, aw[l].kvfT, H_, 256, 256);   // feature rows 0..127 only
        tconv(ps[l]->m_w1, aw[l].mw1T, H_, 256, 256);
        tconv(ps[l]->m_w2, aw[l].mw2T, 256, 256, 256);
        tconv(ps[l]->m_w3, aw[l].mw3T, 256, H_, H_);
    }
    tconv(m1_w1,             WnT,  H_, 256, 256);       // node-feature part
    tconv(m1_w1 + 128 * 256, WaT,  H_, 256, 256);       // anchor-feature part
    tconv(m1_w2, m1w2T, 256, 256, 256);
    tconv(m1_w3, m1w3T, 256, H_, H_);
    tconv(m2_w1, m2w1T, H_, 256, 256);
    tconv(m2_w2, m2w2T, 256, 256, 256);
    tconv(m2_w3, m2w3T, 256, H_, H_);

    // ---- score layer + top-k + gather ----
    maskmul_kernel<<<(BN_ * H_ + 255) / 256, 256, 0, stream>>>(node_f, node_mask, nfm, BN_, H_);
    gemm(nfm, slw1T, sl_b1, hbuf, BN_, H_, H_, true);
    gemm(hbuf, slw2T, sl_b2, svec, BN_, H_, H_, true);
    score_kernel<<<(BN_ + 127) / 128, 128, 0, stream>>>(svec, topk_w, score, BN_);
    topk_kernel<<<B_, 128, 0, stream>>>(score, topi, tops);
    gather_anchor<<<(BK_ * H_ + 255) / 256, 256, 0, stream>>>(node_x, svec, topi, tops, anchor_x, afA);

    // ---- attention layers ----
    auto run_attn = [&](const AttnP& P, const AttnW& W, const float* qf, const float* kvf,
                        const float* kvx, const float* mask, int Kv, float* af_out) {
        gemm(qf,  W.qwT,  P.q_b,  qbuf,   BK_,      H_,  H_, false);
        gemm(kvf, W.kvfT, P.kv_b, kvbase, B_ * Kv,  256, H_, false);
        // t = q @ We_k^T   (We_k = kv_w rows 128..143, cols 0..127)
        sgemm_naive<<<(BK_ * 16 + 255) / 256, 256, 0, stream>>>(
            qbuf, P.kv_w + 128 * 256, tvec, BK_, 16, H_, 256, 1, 0);
        conv_row_bf16<<<(B_ * Kv * H_ + 255) / 256, 256, 0, stream>>>(kvbase, kbaseB, B_ * Kv, H_, 256, 0);
        { dim3 g((H_ * Kv + 255) / 256, B_);
          conv_tr_bf16<<<g, 256, 0, stream>>>(kvbase, vbaseT, Kv, H_, 256, H_); }
        // logits: L[b] = q[b] @ kbase[b]^T
        gemm(qbuf, kbaseB, nullptr, Lbuf, K_, Kv, H_, false,
             B_, (long)K_ * H_, (long)Kv * H_, (long)K_ * Kv);
        attn_softmax_kernel<<<BK_, 128, 0, stream>>>(anchor_x, kvx, tvec, Lbuf, mask, sedge, Kv);
        // upd = attn @ vbase
        gemm(Lbuf, vbaseT, nullptr, upd, K_, H_, Kv, false,
             B_, (long)K_ * Kv, (long)H_ * Kv, (long)K_ * H_);
        // upd += sedge @ We_v  (We_v = kv_w rows 128..143, cols 128..255)
        sgemm_naive<<<(BK_ * H_ + 255) / 256, 256, 0, stream>>>(
            sedge, P.kv_w + 128 * 256 + 128, upd, BK_, H_, 16, 256, 0, 1);
        ln_kernel<<<BK_, 128, 0, stream>>>(qf, upd, nullptr, P.ln1_g, P.ln1_b, fmid);
        gemm(fmid, W.mw1T, P.m_b1, m1b, BK_, 256, H_,  true);
        gemm(m1b,  W.mw2T, P.m_b2, m2b, BK_, 256, 256, true);
        gemm(m2b,  W.mw3T, P.m_b3, m3b, BK_, H_,  256, false);
        ln_kernel<<<BK_, 128, 0, stream>>>(fmid, m3b, nullptr, P.ln2_g, P.ln2_b, af_out);
    };

    run_attn(n2aP,  aw[0], afA,   node_f, node_x,   node_mask, N_, afBuf);  // n2a
    run_attn(a2aP0, aw[1], afBuf, afBuf,  anchor_x, nullptr,   K_, afA);    // a2a[0]
    run_attn(a2aP1, aw[2], afA,   afA,    anchor_x, nullptr,   K_, afBuf);  // a2a[1]
    float* af_final = afBuf;

    // ---- a2n MPNN ----
    gemm(node_f,   WnT, nullptr, nbase, BN_, 256, H_, false);
    gemm(af_final, WaT, nullptr, abase, BK_, 256, H_, false);
    const float* We1 = m1_w1 + 256 * 256;   // edge rows 256..271 of mlp1.w1
    for (int chunk = 0; chunk < BN_ / CN_; ++chunk) {
        int bnBase = chunk * CN_;
        pair_h1_kernel<<<CN_ * K_, 256, 0, stream>>>(nbase, abase, node_x, anchor_x,
                                                     We1, m1_b1, h1, bnBase);
        gemm(h1, m1w2T, m1_b2, h2, CN_ * K_, 256, 256, true);
        gemm(h2, m1w3T, m1_b3, h3, CN_ * K_, H_,  256, false);
        mean_over_k<<<CN_, 128, 0, stream>>>(h3, updN, bnBase);
    }
    ln_kernel<<<BN_, 128, 0, stream>>>(node_f, updN, node_mask, a2n_ln1_g, a2n_ln1_b, nf1);
    gemm(nf1, m2w1T, m2_b1, nm1, BN_, 256, H_,  true);
    gemm(nm1, m2w2T, m2_b2, nm2, BN_, 256, 256, true);
    gemm(nm2, m2w3T, m2_b3, nm3, BN_, H_,  256, false);
    ln_kernel<<<BN_, 128, 0, stream>>>(nf1, nm3, node_mask, a2n_ln2_g, a2n_ln2_b, (float*)d_out);
    // trailing 2 * zeros(B)
    zero_kernel<<<1, 32, 0, stream>>>((float*)d_out + (size_t)BN_ * H_, 2 * B_);
}
